// FiLM_90658169684538
// MI455X (gfx1250) — matrix operand order repaired and checked by reading, not compiled, not run on hardware
//
#include <hip/hip_runtime.h>

// FiLM LSTM (B=256, T=2048, D=H=64) fused persistent kernel for gfx1250.
// 16 workgroups x 512 threads; each WG owns a 16-row batch tile.
// Per step, wave w computes z[:, 16w:16w+16] = bias + x_t@W + h@U with two
// independent 2-deep v_wmma_f32_16x16x32_bf16 chains (f32 accumulate),
// then gates via LDS. Sigmoid uses native v_exp_f32 + v_rcp_f32.

#define B_  256
#define T_  2048
#define D_  64
#define H_  64
#define G4  256   // 4*H

typedef __attribute__((ext_vector_type(16))) __bf16 v16bf;
typedef __attribute__((ext_vector_type(8)))  float  v8f;

union Frag {
    unsigned int u[8];
    v16bf        v;
};

static __device__ __forceinline__ unsigned short f2bf(float f) {
    // round-to-nearest-even f32 -> bf16
    unsigned int u = __float_as_uint(f);
    unsigned int r = u + 0x7FFFu + ((u >> 16) & 1u);
    return (unsigned short)(r >> 16);
}
static __device__ __forceinline__ unsigned int pack_bf(float lo, float hi) {
    return (unsigned int)f2bf(lo) | ((unsigned int)f2bf(hi) << 16);
}
static __device__ __forceinline__ float sigm(float x) {
    // 3 VALU ops: v_exp_f32 (w/ log2e fold), v_add_f32, v_rcp_f32
    return __builtin_amdgcn_rcpf(1.0f + __expf(-x));
}

__global__ __launch_bounds__(512, 1)
void lstm_film_kernel(const float* __restrict__ X,     // [B,T,D]
                      const float* __restrict__ W,     // [D,4H] row-major
                      const float* __restrict__ U,     // [H,4H] row-major
                      const float* __restrict__ bias,  // [4H]
                      const float* __restrict__ wg,    // [H]
                      const float* __restrict__ bg,    // [1]
                      const float* __restrict__ wb,    // [H]
                      const float* __restrict__ bb,    // [1]
                      float* __restrict__ out)         // [2*B]: gamma | beta
{
    __shared__ unsigned short xbuf[2][16 * 64];  // bf16 x tile, double buffered
    __shared__ unsigned short hbf[16 * 64];      // bf16 h (A operand for h@U)
    __shared__ float          hf32[16 * 64];     // f32 h (for the heads)
    __shared__ float          cbuf[16 * 64];     // f32 cell state
    __shared__ float          sg4[16 * 256];     // sigmoid(z) staging

    const int tid  = threadIdx.x;
    const int lane = tid & 31;
    const int wid  = tid >> 5;        // wave id 0..15 -> N tile
    const int half = lane >> 4;       // 0/1 (lane half)
    const int l16  = lane & 15;
    const int n0   = wid << 4;        // N column base in [0,256)
    const int b0   = blockIdx.x << 4; // batch tile base

    // ---- pack W,U B-fragments into VGPRs once (bf16, 2 K-chunks each) ----
    // B layout: lane -> N = n0+l16; VGPR v holds pair K = 32c + half*8 + 2v + (v>=4 ? 8 : 0).
    Frag fW0, fW1, fU0, fU1;
    {
        const int n = n0 + l16;
        #pragma unroll
        for (int v = 0; v < 8; ++v) {
            const int k = (half << 3) + (v << 1) + ((v >> 2) << 3);
            fW0.u[v] = pack_bf(W[(k     ) * G4 + n], W[(k +  1) * G4 + n]);
            fW1.u[v] = pack_bf(W[(k + 32) * G4 + n], W[(k + 33) * G4 + n]);
            fU0.u[v] = pack_bf(U[(k     ) * G4 + n], U[(k +  1) * G4 + n]);
            fU1.u[v] = pack_bf(U[(k + 32) * G4 + n], U[(k + 33) * G4 + n]);
        }
    }
    const float bias_n = bias[n0 + l16];

    // element ownership for gate-combine / staging: 2 elems per thread
    const int prow = tid >> 5;           // 0..15
    const int pcol = (tid & 31) << 1;    // 0,2,...,62
    const int pidx = prow * 64 + pcol;
    const size_t xrow_base = (size_t)(b0 + prow) * (size_t)T_ * D_ + (size_t)pcol;

    // ---- init h0=c0=0, stage x_0 into buffer 0 ----
    {
        hf32[pidx] = 0.0f; hf32[pidx + 1] = 0.0f;
        cbuf[pidx] = 0.0f; cbuf[pidx + 1] = 0.0f;
        *(unsigned int*)&hbf[pidx] = 0u;
        const float2 x0 = *(const float2*)&X[xrow_base];
        *(unsigned int*)&xbuf[0][pidx] = pack_bf(x0.x, x0.y);
    }
    __syncthreads();

    int p = 0;
    for (int t = 0; t < T_; ++t) {
        // prefetch x_{t+1} into registers (overlaps the matrix work)
        const int tn = (t + 1 < T_) ? (t + 1) : t;
        const float2 xn = *(const float2*)&X[xrow_base + (size_t)tn * D_];

        // ---- A fragments (16-bit A 16x32 layout): M=l16, K-octet by half ----
        Frag ax0, ax1, ah0, ah1;
        {
            const int rb = l16 * 64;
            const int o  = half << 3;
            uint4 q;
            q = *(const uint4*)&xbuf[p][rb      + o]; ax0.u[0]=q.x; ax0.u[1]=q.y; ax0.u[2]=q.z; ax0.u[3]=q.w;
            q = *(const uint4*)&xbuf[p][rb + 16 + o]; ax0.u[4]=q.x; ax0.u[5]=q.y; ax0.u[6]=q.z; ax0.u[7]=q.w;
            q = *(const uint4*)&xbuf[p][rb + 32 + o]; ax1.u[0]=q.x; ax1.u[1]=q.y; ax1.u[2]=q.z; ax1.u[3]=q.w;
            q = *(const uint4*)&xbuf[p][rb + 48 + o]; ax1.u[4]=q.x; ax1.u[5]=q.y; ax1.u[6]=q.z; ax1.u[7]=q.w;
            q = *(const uint4*)&hbf[rb      + o];     ah0.u[0]=q.x; ah0.u[1]=q.y; ah0.u[2]=q.z; ah0.u[3]=q.w;
            q = *(const uint4*)&hbf[rb + 16 + o];     ah0.u[4]=q.x; ah0.u[5]=q.y; ah0.u[6]=q.z; ah0.u[7]=q.w;
            q = *(const uint4*)&hbf[rb + 32 + o];     ah1.u[0]=q.x; ah1.u[1]=q.y; ah1.u[2]=q.z; ah1.u[3]=q.w;
            q = *(const uint4*)&hbf[rb + 48 + o];     ah1.u[4]=q.x; ah1.u[5]=q.y; ah1.u[6]=q.z; ah1.u[7]=q.w;
        }

        // ---- z = (bias + x@W) + (h@U): two independent 2-deep WMMA chains ----
        v8f accX, accH;
        #pragma unroll
        for (int r = 0; r < 8; ++r) { accX[r] = bias_n; accH[r] = 0.0f; }
        accX = __builtin_amdgcn_wmma_f32_16x16x32_bf16(false, ax0.v, false, fW0.v, (short)0, accX, false, false);
        accH = __builtin_amdgcn_wmma_f32_16x16x32_bf16(false, ah0.v, false, fU0.v, (short)0, accH, false, false);
        accX = __builtin_amdgcn_wmma_f32_16x16x32_bf16(false, ax1.v, false, fW1.v, (short)0, accX, false, false);
        accH = __builtin_amdgcn_wmma_f32_16x16x32_bf16(false, ah1.v, false, fU1.v, (short)0, accH, false, false);

        // ---- sigmoid(z) -> LDS staging (C layout: VGPR r -> M = r + half*8) ----
        #pragma unroll
        for (int r = 0; r < 8; ++r) {
            const int row = r + (half << 3);
            sg4[row * 256 + n0 + l16] = sigm(accX[r] + accH[r]);
        }

        // stash prefetched x into the other buffer (bf16)
        *(unsigned int*)&xbuf[p ^ 1][pidx] = pack_bf(xn.x, xn.y);

        __syncthreads();

        // ---- gate combine: c = sig(f)*c + sig(i)*sig(g); h = sig(o)*sig(c) ----
        {
            float hn[2];
            #pragma unroll
            for (int e = 0; e < 2; ++e) {
                const int cc = pcol + e;
                const float si = sg4[prow * 256 +       cc];
                const float sf = sg4[prow * 256 +  64 + cc];
                const float sgc= sg4[prow * 256 + 128 + cc];
                const float so = sg4[prow * 256 + 192 + cc];
                const float cn = sf * cbuf[pidx + e] + si * sgc;
                hn[e] = so * sigm(cn);
                cbuf[pidx + e] = cn;
                hf32[pidx + e] = hn[e];
            }
            *(unsigned int*)&hbf[pidx] = pack_bf(hn[0], hn[1]);
        }
        __syncthreads();
        p ^= 1;
    }

    // ---- FiLM heads from final h (f32): gamma | beta ----
    if (tid < 16) {
        float g  = bg[0];
        float bt = bb[0];
        #pragma unroll 8
        for (int j = 0; j < H_; ++j) {
            const float hv = hf32[tid * 64 + j];
            g  += hv * wg[j];
            bt += hv * wb[j];
        }
        out[b0 + tid]      = g;
        out[B_ + b0 + tid] = bt;
    }
}

extern "C" void kernel_launch(void* const* d_in, const int* in_sizes, int n_in,
                              void* d_out, int out_size, void* d_ws, size_t ws_size,
                              hipStream_t stream) {
    (void)in_sizes; (void)n_in; (void)d_ws; (void)ws_size; (void)out_size;
    const float* X    = (const float*)d_in[0];  // lstm_input [B,T,D]
    // d_in[1] inputs_state: only batch size used by reference -> ignored
    const float* W    = (const float*)d_in[2];  // [D,4H]
    const float* U    = (const float*)d_in[3];  // [H,4H]
    const float* bias = (const float*)d_in[4];  // [4H]
    const float* wg   = (const float*)d_in[5];  // [H,1]
    const float* bg   = (const float*)d_in[6];  // [1]
    const float* wb   = (const float*)d_in[7];  // [H,1]
    const float* bb   = (const float*)d_in[8];  // [1]
    float* out = (float*)d_out;                 // [B gamma | B beta]

    lstm_film_kernel<<<dim3(B_ / 16), dim3(512), 0, stream>>>(
        X, W, U, bias, wg, bg, wb, bb, out);
}
